// SCDSC_4337916969109
// MI455X (gfx1250) — matrix operand order repaired and checked by reading, not compiled, not run on hardware
//
#include <hip/hip_runtime.h>
#include <hip/hip_bf16.h>

#define NCELLS 20000
#define DIN    2000
#define KPAD   2016      // DIN padded to a multiple of 32 for the WMMA K-loop
#define NEDGE  640000

typedef __attribute__((ext_vector_type(16))) __bf16         v16bf;
typedef __attribute__((ext_vector_type(8)))  float          v8f;
typedef __attribute__((ext_vector_type(8)))  unsigned short us8;
typedef __attribute__((ext_vector_type(4)))  float          f32x4;

__device__ __forceinline__ unsigned short f2bf(float f) {
  union { float f; unsigned int u; } c; c.f = f;
  unsigned int u = c.u;
  u += 0x7FFFu + ((u >> 16) & 1u);   // round-to-nearest-even
  return (unsigned short)(u >> 16);
}

__device__ __forceinline__ uint2 pk4bf(f32x4 v) {
  uint2 r;
  r.x = (unsigned)f2bf(v.x) | ((unsigned)f2bf(v.y) << 16);
  r.y = (unsigned)f2bf(v.z) | ((unsigned)f2bf(v.w) << 16);
  return r;
}

// ---------------------------------------------------------------------------
// C[M,N] = A[M,K] @ W[K,N] (+ bias[N])   via V_WMMA_F32_16X16X32_BF16
// REQUIRES K % 32 == 0 (caller pads).  block = 256 threads (8 waves),
// tile 128(M) x 64(N), K-step 32; wave w computes rows [w*16, w*16+16).
// ---------------------------------------------------------------------------
__global__ __launch_bounds__(256) void wmma_gemm_kernel(
    const float* __restrict__ A, const float* __restrict__ W,
    const float* __restrict__ bias, float* __restrict__ C,
    int M, int N, int K) {
  __shared__ unsigned short Asm[128 * 32];   // A tile, [m][k]
  __shared__ unsigned short Bsm[64 * 32];    // W tile transposed, [n][k]

  const int tid  = threadIdx.x;
  const int lane = tid & 31;
  const int wave = tid >> 5;
  const int hl   = lane >> 4;     // half-wave select
  const int lm   = lane & 15;

  const int row0 = blockIdx.y * 128;
  const int col0 = blockIdx.x * 64;

  v8f acc[4] = {};

  // A staging: thread covers half a 32-elem row (16 k's)
  const int am = tid >> 1;
  const int ak = (tid & 1) * 16;
  const bool arow_ok = (row0 + am) < M;
  const float* aptr = A + (size_t)(row0 + am) * K + ak;

  // B staging: fixed column per thread, k = bk + 4j  (coalesced along n)
  const int bn = tid & 63;
  const int bk = tid >> 6;                 // 0..3
  const bool bcol_ok = (col0 + bn) < N;
  const float* bptr = W + col0 + bn;

  for (int k0 = 0; k0 < K; k0 += 32) {
    __syncthreads();

    // ---- stage A tile (b128 loads -> packed bf16 -> b64 LDS stores) ----
    if (arow_ok) {
      const f32x4* src = (const f32x4*)(aptr + k0);
#pragma unroll
      for (int j = 0; j < 4; ++j)
        *(uint2*)&Asm[am * 32 + ak + j * 4] = pk4bf(src[j]);
      if ((k0 + 32) < K) __builtin_prefetch(aptr + k0 + 32, 0, 1);
    } else {
      uint2 zz; zz.x = 0u; zz.y = 0u;
#pragma unroll
      for (int j = 0; j < 4; ++j)
        *(uint2*)&Asm[am * 32 + ak + j * 4] = zz;
    }

    // ---- stage B tile (transposed into [n][k]) ----
    if (bcol_ok) {
#pragma unroll
      for (int j = 0; j < 8; ++j)
        Bsm[bn * 32 + bk + 4 * j] = f2bf(bptr[(size_t)(k0 + bk + 4 * j) * N]);
    } else {
#pragma unroll
      for (int j = 0; j < 8; ++j)
        Bsm[bn * 32 + bk + 4 * j] = 0;
    }
    __syncthreads();

    // ---- fragments: issue all ds loads, then 4 back-to-back WMMAs ----
    union { us8 u[2]; v16bf v; } af, bf4[4];
    const us8* ap = (const us8*)&Asm[(wave * 16 + lm) * 32 + hl * 16];
    af.u[0] = ap[0];           // A: lane<16 -> K 0..15; else K 16..31
    af.u[1] = ap[1];
#pragma unroll
    for (int t = 0; t < 4; ++t) {
      const us8* bp = (const us8*)&Bsm[(t * 16 + lm) * 32 + hl * 16];
      bf4[t].u[0] = bp[0];     // B: lane<16 -> K 0..15 of col lm; else K 16..31
      bf4[t].u[1] = bp[1];
    }
#pragma unroll
    for (int t = 0; t < 4; ++t)
      acc[t] = __builtin_amdgcn_wmma_f32_16x16x32_bf16(
          false, af.v, false, bf4[t].v, (short)0, acc[t], false, false);
  }

  // C/D layout: lane = n + 16*(m>=8); VGPR r -> m = r + 8*(lane>=16)
  const int m_base = row0 + wave * 16 + hl * 8;
  const bool rows_ok = (m_base + 8) <= M;
#pragma unroll
  for (int t = 0; t < 4; ++t) {
    int col = col0 + t * 16 + lm;
    if (col < N) {
      float bv = bias ? bias[col] : 0.0f;
      float* cp = C + (size_t)m_base * N + col;
      if (rows_ok) {
#pragma unroll
        for (int r = 0; r < 8; ++r) cp[(size_t)r * N] = acc[t][r] + bv;
      } else {
#pragma unroll
        for (int r = 0; r < 8; ++r)
          if ((m_base + r) < M) cp[(size_t)r * N] = acc[t][r] + bv;
      }
    }
  }
}

// ---------------------------------------------------------------------------
// padding helpers (run once per launch so the GEMM K-loop needs no K guards)
// ---------------------------------------------------------------------------
__global__ void pad_cols_kernel(const float* __restrict__ in, float* __restrict__ outp,
                                int Cin, int Cout) {
  int c = blockIdx.x * blockDim.x + threadIdx.x;
  int r = blockIdx.y;
  if (c < Cout)
    outp[(size_t)r * Cout + c] = (c < Cin) ? in[(size_t)r * Cin + c] : 0.f;
}

__global__ void pad_rows_kernel(const float* __restrict__ in, float* __restrict__ outp,
                                int Rin, int C) {
  int c = blockIdx.x * blockDim.x + threadIdx.x;
  int r = blockIdx.y;
  if (c < C)
    outp[(size_t)r * C + c] = (r < Rin) ? in[(size_t)r * C + c] : 0.f;
}

// ---------------------------------------------------------------------------
// BatchNorm column statistics (biased var, training-mode) -> mean, rstd
// ---------------------------------------------------------------------------
__global__ __launch_bounds__(256) void bn_stats_kernel(
    const float* __restrict__ X, int n, int c,
    float* __restrict__ mean, float* __restrict__ rstd) {
  __shared__ float ss[256], ss2[256];
  int col = blockIdx.x;
  float s = 0.f, s2 = 0.f;
  for (int r = threadIdx.x; r < n; r += 256) {
    float v = X[(size_t)r * c + col];
    s += v; s2 += v * v;
  }
  ss[threadIdx.x] = s; ss2[threadIdx.x] = s2;
  __syncthreads();
  for (int w = 128; w > 0; w >>= 1) {
    if (threadIdx.x < w) {
      ss[threadIdx.x]  += ss[threadIdx.x + w];
      ss2[threadIdx.x] += ss2[threadIdx.x + w];
    }
    __syncthreads();
  }
  if (threadIdx.x == 0) {
    float m   = ss[0] / n;
    float var = ss2[0] / n - m * m;
    mean[col] = m;
    rstd[col] = rsqrtf(var + 1e-5f);
  }
}

// c is always a power of two here -> column via mask
__global__ void bn_apply_kernel(float* __restrict__ Y, const float* __restrict__ mean,
                                const float* __restrict__ rstd,
                                const float* __restrict__ g, const float* __restrict__ b,
                                int cmask, int total, int relu) {
  int i = blockIdx.x * blockDim.x + threadIdx.x;
  int st = gridDim.x * blockDim.x;
  for (; i < total; i += st) {
    int col = i & cmask;
    float v = (Y[i] - mean[col]) * rstd[col] * g[col] + b[col];
    if (relu) v = fmaxf(v, 0.f);
    Y[i] = v;
  }
}

__global__ void zero4_kernel(f32x4* __restrict__ Y, int n4) {
  int i = blockIdx.x * blockDim.x + threadIdx.x;
  int st = gridDim.x * blockDim.x;
  f32x4 z = {0.f, 0.f, 0.f, 0.f};
  for (; i < n4; i += st) Y[i] = z;
}

__global__ void relu4_kernel(f32x4* __restrict__ Y, int n4) {
  int i = blockIdx.x * blockDim.x + threadIdx.x;
  int st = gridDim.x * blockDim.x;
  for (; i < n4; i += st) {
    f32x4 v = Y[i];
    v.x = fmaxf(v.x, 0.f); v.y = fmaxf(v.y, 0.f);
    v.z = fmaxf(v.z, 0.f); v.w = fmaxf(v.w, 0.f);
    Y[i] = v;
  }
}

__global__ void mix4_kernel(const f32x4* __restrict__ H, const f32x4* __restrict__ F,
                            f32x4* __restrict__ O, int n4) {
  int i = blockIdx.x * blockDim.x + threadIdx.x;
  int st = gridDim.x * blockDim.x;
  for (; i < n4; i += st) {
    f32x4 h = H[i], f = F[i];
    f32x4 r;
    r.x = 0.5f * h.x + 0.5f * f.x; r.y = 0.5f * h.y + 0.5f * f.y;   // sigma=0.5
    r.z = 0.5f * h.z + 0.5f * f.z; r.w = 0.5f * h.w + 0.5f * f.w;
    O[i] = r;
  }
}

// out[rows[e], :] += vals[e] * sup[cols[e], :]   (segment_sum over edges)
__global__ void spmm_kernel(const int* __restrict__ rows, const int* __restrict__ cols,
                            const float* __restrict__ vals, const float* __restrict__ sup,
                            float* __restrict__ out, int E, int F) {
  for (int e = blockIdx.x; e < E; e += gridDim.x) {
    int r = rows[e], c = cols[e];
    float v = vals[e];
    const float* sp = sup + (size_t)c * F;
    float* op = out + (size_t)r * F;
    for (int f = threadIdx.x; f < F; f += blockDim.x)
      atomicAdd(op + f, v * sp[f]);
  }
}

__global__ void softmax16_kernel(const float* __restrict__ H, float* __restrict__ P, int n) {
  int r = blockIdx.x * blockDim.x + threadIdx.x;
  if (r >= n) return;
  const float* p = H + (size_t)r * 16;
  float mx = p[0];
#pragma unroll
  for (int j = 1; j < 16; ++j) mx = fmaxf(mx, p[j]);
  float e[16]; float s = 0.f;
#pragma unroll
  for (int j = 0; j < 16; ++j) { e[j] = expf(p[j] - mx); s += e[j]; }
  float inv = 1.f / s;
#pragma unroll
  for (int j = 0; j < 16; ++j) P[(size_t)r * 16 + j] = e[j] * inv;
}

// mode 0: clip(exp(x)); 1: clip(softplus(x)); 2: sigmoid(x)
__global__ void head_transform_kernel(float* __restrict__ Y, int n, int mode) {
  int i = blockIdx.x * blockDim.x + threadIdx.x;
  int st = gridDim.x * blockDim.x;
  for (; i < n; i += st) {
    float v = Y[i];
    if (mode == 0)      v = fminf(fmaxf(expf(v), 1e-5f), 1e6f);
    else if (mode == 1) { float sp = v > 20.f ? v : log1pf(expf(v));
                          v = fminf(fmaxf(sp, 1e-4f), 1e4f); }
    else                v = 1.f / (1.f + expf(-v));
    Y[i] = v;
  }
}

// Student-t q, V=1 -> exponent (V+1)/2 = 1, just normalize 1/(1+dist)
__global__ void studentt_kernel(const float* __restrict__ Z, const float* __restrict__ Cl,
                                float* __restrict__ Q, int n) {
  int r = blockIdx.x * blockDim.x + threadIdx.x;
  if (r >= n) return;
  float z[32];
#pragma unroll
  for (int j = 0; j < 32; ++j) z[j] = Z[(size_t)r * 32 + j];
  float qv[16]; float s = 0.f;
  for (int k = 0; k < 16; ++k) {
    float d = 0.f;
#pragma unroll
    for (int j = 0; j < 32; ++j) { float t = z[j] - Cl[k * 32 + j]; d += t * t; }
    float q = 1.f / (1.f + d);
    qv[k] = q; s += q;
  }
  float inv = 1.f / s;
  for (int k = 0; k < 16; ++k) Q[(size_t)r * 16 + k] = qv[k] * inv;
}

// ---------------------------------------------------------------------------
extern "C" void kernel_launch(void* const* d_in, const int* in_sizes, int n_in,
                              void* d_out, int out_size, void* d_ws, size_t ws_size,
                              hipStream_t stream) {
  (void)in_sizes; (void)n_in; (void)out_size; (void)ws_size;

  // -- inputs, setup_inputs() insertion order --
  const float* x        = (const float*)d_in[0];
  const float* adj_vals = (const float*)d_in[1];
  const float* LW[10]; const float* Lb[10];
  for (int i = 0; i < 10; ++i) { LW[i] = (const float*)d_in[2 + 2*i];
                                 Lb[i] = (const float*)d_in[3 + 2*i]; }
  const float* Bg[9]; const float* Bb[9];
  for (int i = 0; i < 9; ++i)  { Bg[i] = (const float*)d_in[22 + 2*i];
                                 Bb[i] = (const float*)d_in[23 + 2*i]; }
  const float* GW[7];
  for (int i = 0; i < 7; ++i)  GW[i] = (const float*)d_in[40 + i];
  const float* cluster = (const float*)d_in[47];
  const float* HW[3]; const float* Hb[3];
  for (int i = 0; i < 3; ++i)  { HW[i] = (const float*)d_in[48 + 2*i];
                                 Hb[i] = (const float*)d_in[49 + 2*i]; }
  const int* arows = (const int*)d_in[54];
  const int* acols = (const int*)d_in[55];

  // -- output slices (return order: x_bar, q, predict, z3, mean, disp, pi) --
  float* out  = (float*)d_out;
  float* xbar = out;
  float* qout = xbar + (size_t)NCELLS * DIN;
  float* pred = qout + (size_t)NCELLS * 16;
  float* z3   = pred + (size_t)NCELLS * 16;
  float* mnO  = z3   + (size_t)NCELLS * 32;
  float* dsO  = mnO  + (size_t)NCELLS * DIN;
  float* piO  = dsO  + (size_t)NCELLS * DIN;

  // -- workspace --
  float* ws = (float*)d_ws;
  size_t o = 0;
  auto alloc = [&](size_t n) { float* p = ws + o; o += n; return p; };
  float* h1    = alloc((size_t)NCELLS * 1024);
  float* h2    = alloc((size_t)NCELLS * 512);
  float* h3    = alloc((size_t)NCELLS * 256);
  float* z1    = alloc((size_t)NCELLS * 128);
  float* z2    = alloc((size_t)NCELLS * 64);
  float* d1    = alloc((size_t)NCELLS * 256);
  float* d2    = alloc((size_t)NCELLS * 512);
  float* d3    = alloc((size_t)NCELLS * 1024);
  float* gh    = alloc((size_t)NCELLS * 1024);
  float* gmix  = alloc((size_t)NCELLS * 1024);
  float* gsup  = alloc((size_t)NCELLS * 1024);
  float* xpad  = alloc((size_t)NCELLS * KPAD);   // x zero-padded to K%32==0
  float* w0pad = alloc((size_t)KPAD * 1024);     // LW[0] zero-padded rows
  float* g0pad = alloc((size_t)KPAD * 1024);     // GW[0] zero-padded rows
  float* mbuf  = alloc(2048);
  float* rbuf  = alloc(2048);

  auto ewb = [](int n) { return (n + 255) / 256; };
  auto gemm = [&](const float* A, const float* Wm, const float* bias, float* Cm,
                  int M, int N, int K) {
    dim3 g((N + 63) / 64, (M + 127) / 128);
    wmma_gemm_kernel<<<g, 256, 0, stream>>>(A, Wm, bias, Cm, M, N, K);
  };
  auto ae_bn = [&](float* op, int outD, int li, bool relu) {
    bn_stats_kernel<<<outD, 256, 0, stream>>>(op, NCELLS, outD, mbuf, rbuf);
    int tot = NCELLS * outD;
    bn_apply_kernel<<<ewb(tot), 256, 0, stream>>>(op, mbuf, rbuf, Bg[li], Bb[li],
                                                  outD - 1, tot, relu ? 1 : 0);
  };
  auto ae_layer = [&](const float* in, float* op, int inD, int outD, int li, bool relu) {
    gemm(in, LW[li], Lb[li], op, NCELLS, outD, inD);
    ae_bn(op, outD, li, relu);
  };

  // ---------------- K padding pre-pass ----------------
  pad_cols_kernel<<<dim3((KPAD + 255) / 256, NCELLS), 256, 0, stream>>>(x, xpad, DIN, KPAD);
  pad_rows_kernel<<<dim3(4, KPAD), 256, 0, stream>>>(LW[0], w0pad, DIN, 1024);
  pad_rows_kernel<<<dim3(4, KPAD), 256, 0, stream>>>(GW[0], g0pad, DIN, 1024);

  // ---------------- autoencoder ----------------
  gemm(xpad, w0pad, Lb[0], h1, NCELLS, 1024, KPAD);
  ae_bn(h1, 1024, 0, true);
  ae_layer(h1, h2, 1024, 512, 1, true);
  ae_layer(h2, h3,  512, 256, 2, true);
  ae_layer(h3, z1,  256, 128, 3, false);
  ae_layer(z1, z2,  128,  64, 4, false);
  ae_layer(z2, z3,   64,  32, 5, false);     // z3 straight into d_out slice
  ae_layer(z3, d1,   32, 256, 6, true);
  ae_layer(d1, d2,  256, 512, 7, true);
  ae_layer(d2, d3,  512, 1024, 8, true);
  gemm(d3, LW[9], Lb[9], xbar, NCELLS, DIN, 1024);   // x_bar

  // ---------------- GCN stack ----------------
  const int gin[7]  = {KPAD, 1024, 512, 256, 128, 64, 32};
  const int gout[7] = {1024, 512, 256, 128, 64, 32, 16};
  const float* feats[6] = {h1, h2, h3, z1, z2, z3};

  for (int i = 0; i < 7; ++i) {
    const float* gi;
    const float* gw = (i == 0) ? g0pad : GW[i];
    if (i == 0) {
      gi = xpad;
    } else {
      int tin4 = (NCELLS * gin[i]) / 4;
      mix4_kernel<<<ewb(tin4), 256, 0, stream>>>((const f32x4*)gh, (const f32x4*)feats[i - 1],
                                                 (f32x4*)gmix, tin4);
      gi = gmix;
    }
    gemm(gi, gw, nullptr, gsup, NCELLS, gout[i], gin[i]);
    int tot4 = (NCELLS * gout[i]) / 4;
    zero4_kernel<<<ewb(tot4), 256, 0, stream>>>((f32x4*)gh, tot4);
    int tpb = gout[i] >= 256 ? 256 : 64;
    spmm_kernel<<<65536, tpb, 0, stream>>>(arows, acols, adj_vals, gsup, gh,
                                           NEDGE, gout[i]);
    if (i < 6) relu4_kernel<<<ewb(tot4), 256, 0, stream>>>((f32x4*)gh, tot4);
  }
  softmax16_kernel<<<(NCELLS + 255) / 256, 256, 0, stream>>>(gh, pred, NCELLS);

  // ---------------- ZINB heads ----------------
  int htot = NCELLS * DIN;
  gemm(d3, HW[0], Hb[0], mnO, NCELLS, DIN, 1024);
  head_transform_kernel<<<ewb(htot), 256, 0, stream>>>(mnO, htot, 0);
  gemm(d3, HW[1], Hb[1], dsO, NCELLS, DIN, 1024);
  head_transform_kernel<<<ewb(htot), 256, 0, stream>>>(dsO, htot, 1);
  gemm(d3, HW[2], Hb[2], piO, NCELLS, DIN, 1024);
  head_transform_kernel<<<ewb(htot), 256, 0, stream>>>(piO, htot, 2);

  // ---------------- Student-t q ----------------
  studentt_kernel<<<(NCELLS + 255) / 256, 256, 0, stream>>>(z3, cluster, qout, NCELLS);
}
